// MemN2N_37503654429128
// MI455X (gfx1250) — matrix operand order repaired and checked by reading, not compiled, not run on hardware
//
#include <hip/hip_runtime.h>
#include <hip/hip_bf16.h>

#define MM      4096
#define KK      32000
#define EE      128
#define KSPLIT  8
#define KCHUNK  (KK / KSPLIT)   // 4000
#define KSTEP   32
#define LDSP    40              // 32 k + 4-DWORD pad -> conflict-free, 16B-aligned rows

typedef __attribute__((ext_vector_type(16))) __bf16 v16bf;
typedef __attribute__((ext_vector_type(8)))  __bf16 v8bf;
typedef __attribute__((ext_vector_type(8)))  float  v8f;
typedef __attribute__((ext_vector_type(4)))  float  v4f;
typedef __attribute__((ext_vector_type(4)))  unsigned int v4u;
typedef __attribute__((ext_vector_type(8)))  int    v8i;
typedef __attribute__((ext_vector_type(4)))  int    v4i;

#if defined(__gfx1250__) && __has_builtin(__builtin_amdgcn_tensor_load_to_lds)
#define HAS_TDM 1
#else
#define HAS_TDM 0
#endif

// ---------------------------------------------------------------------------
// Zero small accumulation slots (o, max, sum)
// ---------------------------------------------------------------------------
__global__ void init_ws(float* __restrict__ w, int n) {
    int i = blockIdx.x * blockDim.x + threadIdx.x;
    if (i < n) w[i] = 0.0f;
}

// ---------------------------------------------------------------------------
// One-time fp32 -> bf16 conversion of the A and C embedding matrices
// ---------------------------------------------------------------------------
__global__ void convert_embed(const float* __restrict__ a, const float* __restrict__ c,
                              __bf16* __restrict__ ab, __bf16* __restrict__ cb) {
    const int n4 = (EE * KK) / 4;
    const int stride = gridDim.x * blockDim.x;
    for (int i = blockIdx.x * blockDim.x + threadIdx.x; i < n4; i += stride) {
        v4f fa = ((const v4f*)a)[i];
        v4f fc = ((const v4f*)c)[i];
        typedef __attribute__((ext_vector_type(4))) __bf16 v4bf;
        v4bf ba, bc;
#pragma unroll
        for (int q = 0; q < 4; ++q) { ba[q] = (__bf16)fa[q]; bc[q] = (__bf16)fc[q]; }
        ((v4bf*)ab)[i] = ba;
        ((v4bf*)cb)[i] = bc;
    }
}

#if HAS_TDM
// ---------------------------------------------------------------------------
// TDM: 2D tile load global->LDS with hardware row padding.
//   tile: tile_d1 rows x tile_d0 elements, element size 2^dsz bytes
//   pad:  pad_int code (interval = 2^(code+1) DWORDs), pad_amt code (amt = code+1 DWORDs)
// ---------------------------------------------------------------------------
__device__ __forceinline__ void tdm_load_2d(const void* gaddr, unsigned lds_off,
                                            unsigned tile_d0, unsigned tile_d1,
                                            unsigned tens_d0, unsigned tens_d1,
                                            unsigned stride0, unsigned dsz,
                                            unsigned pad_en, unsigned pad_int,
                                            unsigned pad_amt) {
    unsigned long long ga = (unsigned long long)(uintptr_t)gaddr;
    v4u g0;
    g0[0] = 1u;                                    // count=1, user descriptor
    g0[1] = lds_off;                               // LDS byte address
    g0[2] = (unsigned)(ga & 0xffffffffull);        // global addr [31:0]
    g0[3] = (unsigned)((ga >> 32) & 0x01ffffffull) // global addr [56:32]
            | 0x80000000u;                         // type=2 ("image")
    v8i g1;
    g1[0] = (int)((dsz << 16) | (pad_en << 20) | (pad_int << 22) | (pad_amt << 25));
    g1[1] = (int)((tens_d0 & 0xffffu) << 16);                        // tensor_dim0 lo
    g1[2] = (int)(((tens_d0 >> 16) & 0xffffu) | ((tens_d1 & 0xffffu) << 16));
    g1[3] = (int)(((tens_d1 >> 16) & 0xffffu) | ((tile_d0 & 0xffffu) << 16));
    g1[4] = (int)(tile_d1 & 0xffffu);                                // tile_dim1, tile_dim2=0
    g1[5] = (int)stride0;                                            // tensor_dim0_stride lo
    g1[6] = 0;
    g1[7] = 0;
    v4i z = {0, 0, 0, 0};
#if __clang_major__ >= 23
    v8i z8 = {0, 0, 0, 0, 0, 0, 0, 0};
    __builtin_amdgcn_tensor_load_to_lds(g0, g1, z, z, z8, 0);
#else
    __builtin_amdgcn_tensor_load_to_lds(g0, g1, z, z, 0);
#endif
}
#endif

__device__ __forceinline__ void stage_fallback(__bf16 (*dst)[LDSP],
                                               const __bf16* __restrict__ src,
                                               int k0, int tid) {
    const int e = tid >> 1, sh = tid & 1;
    const uint4* s = (const uint4*)(src + (size_t)e * KK + k0 + sh * 16);
    uint4 w0 = s[0], w1 = s[1];
    *(uint4*)&dst[e][sh * 16] = w0;
    *(uint4*)&dst[e][sh * 16 + 8] = w1;
}

// ---------------------------------------------------------------------------
// Fused m = mem @ A^T and c = mem @ C^T, bf16 WMMA.
// grid (32, KSPLIT), block 256. TDM double-buffers embed tiles into padded LDS;
// mem A-operands are register-pipelined global loads. Partials -> slabs.
// ---------------------------------------------------------------------------
__global__ __launch_bounds__(256) void gemm_mc(const float* __restrict__ mem,
                                               const __bf16* __restrict__ Ab,
                                               const __bf16* __restrict__ Cb,
                                               float* __restrict__ pm,
                                               float* __restrict__ pc) {
    __shared__ __bf16 lA[2][EE][LDSP];
    __shared__ __bf16 lC[2][EE][LDSP];

    const int tid  = threadIdx.x;
    const int wave = tid >> 5;
    const int lane = tid & 31;
    const int row0 = blockIdx.x * 128 + wave * 16;
    const int k0b  = blockIdx.y * KCHUNK;
    const int er   = lane & 15;
    const int hi   = lane >> 4;

    v8f acc[16];
    const v8f vzero = {0.f, 0.f, 0.f, 0.f, 0.f, 0.f, 0.f, 0.f};
#pragma unroll
    for (int t = 0; t < 16; ++t) acc[t] = vzero;

    const float* memrow = mem + (size_t)(row0 + er) * KK;

    // ---- prologue: stage buffer 0 ----
#if HAS_TDM
    if (wave == 0) {
        tdm_load_2d(Ab + k0b, (unsigned)(uintptr_t)&lA[0][0][0],
                    KSTEP, EE, KK, EE, KK, /*dsz=*/1, /*pad*/1, /*int:16dw*/3, /*amt:4dw*/3);
        tdm_load_2d(Cb + k0b, (unsigned)(uintptr_t)&lC[0][0][0],
                    KSTEP, EE, KK, EE, KK, 1, 1, 3, 3);
    }
#else
    stage_fallback(lA[0], Ab, k0b, tid);
    stage_fallback(lC[0], Cb, k0b, tid);
#endif

    // ---- prologue: mem loads for kk = 0 ----
    v4f f0, f1, f2, f3;
    {
        const v4f* p  = (const v4f*)(memrow + k0b + hi * 8);
        const v4f* p2 = (const v4f*)(memrow + k0b + hi * 8 + 16);
        f0 = p[0]; f1 = p[1]; f2 = p2[0]; f3 = p2[1];
    }

#if HAS_TDM
    __builtin_amdgcn_s_wait_tensorcnt(0);
#endif
    __syncthreads();

    int cur = 0;
    for (int kk = 0; kk < KCHUNK; kk += KSTEP) {
        const bool more = (kk + KSTEP) < KCHUNK;

        // prefetch next embed tiles into the other buffer (async, TENSORcnt)
#if HAS_TDM
        if (more && wave == 0) {
            const int kn = k0b + kk + KSTEP;
            tdm_load_2d(Ab + kn, (unsigned)(uintptr_t)&lA[cur ^ 1][0][0],
                        KSTEP, EE, KK, EE, KK, 1, 1, 3, 3);
            tdm_load_2d(Cb + kn, (unsigned)(uintptr_t)&lC[cur ^ 1][0][0],
                        KSTEP, EE, KK, EE, KK, 1, 1, 3, 3);
        }
#else
        if (more) {
            const int kn = k0b + kk + KSTEP;
            stage_fallback(lA[cur ^ 1], Ab, kn, tid);
            stage_fallback(lC[cur ^ 1], Cb, kn, tid);
        }
#endif

        // consume current mem registers -> bf16 A-operand
        v16bf a;
#pragma unroll
        for (int q = 0; q < 4; ++q) {
            a[q]      = (__bf16)f0[q];
            a[4 + q]  = (__bf16)f1[q];
            a[8 + q]  = (__bf16)f2[q];
            a[12 + q] = (__bf16)f3[q];
        }

        // issue next step's mem loads immediately (latency hidden behind WMMA block)
        if (more) {
            const int kn = k0b + kk + KSTEP + hi * 8;
            const v4f* p  = (const v4f*)(memrow + kn);
            const v4f* p2 = (const v4f*)(memrow + kn + 16);
            f0 = p[0]; f1 = p[1]; f2 = p2[0]; f3 = p2[1];
            if (kk + 2 * KSTEP < KCHUNK)
                __builtin_prefetch(memrow + kn + KSTEP, 0, 1);
        }

        const int koff = hi * 8;
#pragma unroll
        for (int t = 0; t < 8; ++t) {
            v8bf b0 = *(const v8bf*)&lA[cur][t * 16 + er][koff];
            v8bf b1 = *(const v8bf*)&lA[cur][t * 16 + er][koff + 16];
            v16bf b = __builtin_shufflevector(b0, b1, 0, 1, 2, 3, 4, 5, 6, 7,
                                              8, 9, 10, 11, 12, 13, 14, 15);
            acc[t] = __builtin_amdgcn_wmma_f32_16x16x32_bf16(
                false, a, false, b, (short)0, acc[t], false, false);
        }
#pragma unroll
        for (int t = 0; t < 8; ++t) {
            v8bf b0 = *(const v8bf*)&lC[cur][t * 16 + er][koff];
            v8bf b1 = *(const v8bf*)&lC[cur][t * 16 + er][koff + 16];
            v16bf b = __builtin_shufflevector(b0, b1, 0, 1, 2, 3, 4, 5, 6, 7,
                                              8, 9, 10, 11, 12, 13, 14, 15);
            acc[8 + t] = __builtin_amdgcn_wmma_f32_16x16x32_bf16(
                false, a, false, b, (short)0, acc[8 + t], false, false);
        }

#if HAS_TDM
        __builtin_amdgcn_s_wait_tensorcnt(0);
#endif
        __syncthreads();
        cur ^= 1;
    }

    // ---- epilogue: plain stores into this K-slab's partial buffers ----
    float* pmS = pm + (size_t)blockIdx.y * MM * EE;
    float* pcS = pc + (size_t)blockIdx.y * MM * EE;
#pragma unroll
    for (int t = 0; t < 8; ++t) {
#pragma unroll
        for (int j = 0; j < 8; ++j) {
            const int r = row0 + hi * 8 + j;
            pmS[(size_t)r * EE + t * 16 + er] = acc[t][j];
            pcS[(size_t)r * EE + t * 16 + er] = acc[8 + t][j];
        }
    }
}

// ---------------------------------------------------------------------------
// Sum the KSPLIT partial slabs into m and c
// ---------------------------------------------------------------------------
__global__ void reduce_mc(const float* __restrict__ pm, const float* __restrict__ pc,
                          float* __restrict__ m, float* __restrict__ c) {
    const int i = blockIdx.x * blockDim.x + threadIdx.x;   // [0, MM*EE)
    float sm = 0.f, sc = 0.f;
#pragma unroll
    for (int s = 0; s < KSPLIT; ++s) {
        sm += pm[(size_t)s * MM * EE + i];
        sc += pc[(size_t)s * MM * EE + i];
    }
    m[i] = sm;
    c[i] = sc;
}

// ---------------------------------------------------------------------------
// u0 = query @ B^T
// ---------------------------------------------------------------------------
__global__ void embed_query(const float* __restrict__ q,
                            const float* __restrict__ Bm,
                            float* __restrict__ u0) {
    __shared__ float red[256];
    const int e = blockIdx.x, tid = threadIdx.x;
    const float* row = Bm + (size_t)e * KK;
    float acc = 0.f;
    for (int v = tid; v < KK; v += 256) acc += q[v] * row[v];
    red[tid] = acc;
    __syncthreads();
    for (int s = 128; s > 0; s >>= 1) {
        if (tid < s) red[tid] += red[tid + s];
        __syncthreads();
    }
    if (tid == 0) u0[e] = red[0];
}

// ---------------------------------------------------------------------------
// Hop stage A: logits + atomic max (logits positive -> int-order valid)
// ---------------------------------------------------------------------------
__global__ void hopA(const float* __restrict__ m, const float* __restrict__ u,
                     float* __restrict__ l, int* __restrict__ maxslot) {
    __shared__ float su[128];
    __shared__ float red[128];
    const int tid = threadIdx.x;
    su[tid] = u[tid];
    __syncthreads();
    const int i = blockIdx.x * 128 + tid;
    const float* row = m + (size_t)i * EE;
    float acc = 0.f;
#pragma unroll
    for (int e = 0; e < EE; e += 4) {
        v4f f = *(const v4f*)(row + e);
        acc += f[0] * su[e] + f[1] * su[e + 1] + f[2] * su[e + 2] + f[3] * su[e + 3];
    }
    l[i] = acc;
    red[tid] = acc;
    __syncthreads();
    for (int s = 64; s > 0; s >>= 1) {
        if (tid < s) red[tid] = fmaxf(red[tid], red[tid + s]);
        __syncthreads();
    }
    if (tid == 0) atomicMax(maxslot, __float_as_int(red[0]));
}

// ---------------------------------------------------------------------------
// Hop stage B: p[i] = exp(l[i]-max); atomic sum
// ---------------------------------------------------------------------------
__global__ void hopB(const float* __restrict__ l, const float* __restrict__ maxslot,
                     float* __restrict__ p, float* __restrict__ sumslot) {
    __shared__ float red[256];
    const int tid = threadIdx.x;
    const int i = blockIdx.x * 256 + tid;
    const float mx = maxslot[0];
    const float v = __expf(l[i] - mx);
    p[i] = v;
    red[tid] = v;
    __syncthreads();
    for (int s = 128; s > 0; s >>= 1) {
        if (tid < s) red[tid] += red[tid + s];
        __syncthreads();
    }
    if (tid == 0) atomicAdd(sumslot, red[0]);
}

// ---------------------------------------------------------------------------
// Hop stage C: o[e] += sum_i p[i]*c[i,e]
// ---------------------------------------------------------------------------
__global__ void hopC(const float* __restrict__ p, const float* __restrict__ c,
                     float* __restrict__ o) {
    const int e = threadIdx.x;
    const int i0 = blockIdx.x * 512;
    float acc = 0.f;
    for (int i = i0; i < i0 + 512; ++i) acc += p[i] * c[(size_t)i * EE + e];
    atomicAdd(&o[e], acc);
}

// ---------------------------------------------------------------------------
// Hop stage D: u_out = u_in + o / sum
// ---------------------------------------------------------------------------
__global__ void hopD(const float* __restrict__ u_in, const float* __restrict__ o,
                     const float* __restrict__ sumslot, float* __restrict__ u_out) {
    const int e = threadIdx.x;
    u_out[e] = u_in[e] + o[e] / sumslot[0];
}

// ---------------------------------------------------------------------------
extern "C" void kernel_launch(void* const* d_in, const int* in_sizes, int n_in,
                              void* d_out, int out_size, void* d_ws, size_t ws_size,
                              hipStream_t stream) {
    const float* mem   = (const float*)d_in[0];  // (1, 4096, 32000)
    const float* query = (const float*)d_in[1];  // (1, 32000)
    const float* Ae    = (const float*)d_in[2];  // (128, 32000)
    const float* Bm    = (const float*)d_in[3];  // (128, 32000)
    const float* Ce    = (const float*)d_in[4];  // (128, 32000)
    float* out = (float*)d_out;                  // (1, 128)

    // workspace layout: bf16 embeds first (16B-aligned), then float scratch
    __bf16* WAb = (__bf16*)d_ws;                 // EE*KK
    __bf16* WCb = WAb + (size_t)EE * KK;         // EE*KK
    float*  F   = (float*)(WCb + (size_t)EE * KK);
    float* Wpm  = F;                             // KSPLIT*MM*EE
    float* Wpc  = Wpm + (size_t)KSPLIT * MM * EE;
    float* Wm   = Wpc + (size_t)KSPLIT * MM * EE;
    float* Wc   = Wm + MM * EE;
    float* Wo   = Wc + MM * EE;                  // 3*128
    float* Wmax = Wo + 3 * EE;                   // 3
    float* Wsum = Wmax + 3;                      // 3
    float* Wl   = Wsum + 3;                      // 4096
    float* Wp   = Wl + MM;                       // 4096
    float* Wu   = Wp + MM;                       // 3*128

    init_ws<<<2, 256, 0, stream>>>(Wo, 3 * EE + 6);
    convert_embed<<<2048, 256, 0, stream>>>(Ae, Ce, WAb, WCb);
    gemm_mc<<<dim3(32, KSPLIT), 256, 0, stream>>>(mem, WAb, WCb, Wpm, Wpc);
    reduce_mc<<<(MM * EE) / 256, 256, 0, stream>>>(Wpm, Wpc, Wm, Wc);
    embed_query<<<128, 256, 0, stream>>>(query, Bm, Wu);

    for (int h = 0; h < 3; ++h) {
        float* u_in  = Wu + h * EE;
        float* u_out = (h == 2) ? out : (Wu + (h + 1) * EE);
        hopA<<<32, 128, 0, stream>>>(Wm, u_in, Wl, (int*)(Wmax + h));
        hopB<<<16, 256, 0, stream>>>(Wl, Wmax + h, Wp, Wsum + h);
        hopC<<<8, 128, 0, stream>>>(Wp, Wc, Wo + h * EE);
        hopD<<<1, 128, 0, stream>>>(u_in, Wo + h * EE, Wsum + h, u_out);
    }
}
